// LukaQwenAttention_17806934409676
// MI455X (gfx1250) — compile-verified, matrix operand order checked by reading, not run on hardware
//
#include <hip/hip_runtime.h>

typedef __attribute__((ext_vector_type(16))) __bf16 v16bf;
typedef __attribute__((ext_vector_type(8)))  __bf16 v8bf;
typedef __attribute__((ext_vector_type(8)))  float  v8f;
typedef unsigned int u32;
typedef __attribute__((ext_vector_type(4))) u32 v4u;
typedef __attribute__((ext_vector_type(8))) int v8i;
typedef __attribute__((ext_vector_type(4))) int v4i;

#define WMMA_BF16(a,b,c) __builtin_amdgcn_wmma_f32_16x16x32_bf16(false,(a),false,(b),(short)0,(c),false,false)

#if __has_builtin(__builtin_amdgcn_tensor_load_to_lds) && __has_builtin(__builtin_amdgcn_s_wait_tensorcnt)
#define USE_TDM 1
#else
#define USE_TDM 0
#endif

constexpr int Bc  = 2;
constexpr int Sc  = 2048;
constexpr int HIDc= 2048;
constexpr int Hc  = 16;
constexpr int KVc = 8;
constexpr int Dc  = 128;
constexpr float EPSc   = 1e-6f;
constexpr float SCALEc = 0.08838834764831845f; // 128^-0.5

// -------- helpers --------
__device__ inline v16bf ld_frag(const __bf16* p) {
    v8bf lo = *(const v8bf*)p;
    v8bf hi = *(const v8bf*)(p + 8);
    v16bf r;
#pragma unroll
    for (int i = 0; i < 8; ++i) { r[i] = lo[i]; r[i + 8] = hi[i]; }
    return r;
}

__device__ inline void st8bf(__bf16* p, float4 a, float4 b) {
    v8bf v;
    v[0]=(__bf16)a.x; v[1]=(__bf16)a.y; v[2]=(__bf16)a.z; v[3]=(__bf16)a.w;
    v[4]=(__bf16)b.x; v[5]=(__bf16)b.y; v[6]=(__bf16)b.z; v[7]=(__bf16)b.w;
    *(v8bf*)p = v;
}

#if USE_TDM
// TDM: async 2-D bf16 tile (tile0 x tile1 elements) from global -> LDS, with
// row padding so LDS row stride = tile0*2 + 16 bytes. ISA 08_async_tensor §8.
// 6-arg builtin form: (g0, g1, g2, g3, g4-unused, cpol).
__device__ inline void tdm_load_tile_2d(u32 lds_off, const void* gptr,
                                        u32 dim0, u32 dim1,
                                        u32 tile0, u32 tile1, u32 stride0,
                                        u32 pad_int_code, u32 pad_amt_code)
{
    unsigned long long ga = (unsigned long long)(uintptr_t)gptr;
    v4u g0;
    g0.x = 1u;                                    // count=1, user descriptor
    g0.y = lds_off;                               // lds_addr
    g0.z = (u32)ga;                               // global_addr[31:0]
    g0.w = (u32)(ga >> 32) | (2u << 30);          // global_addr[56:32] | type=2
    v8i g1;
    g1[0] = (int)((1u << 16)                      // data_size = 2 bytes
                | (1u << 20)                      // pad_enable
                | (pad_int_code << 22)            // pad interval (DWORDs = 2^(c+1))
                | (pad_amt_code << 25));          // pad amount (DWORDs = c+1)
    g1[1] = (int)(dim0 << 16);                    // tensor_dim0[15:0]
    g1[2] = (int)((dim0 >> 16) | (dim1 << 16));   // tensor_dim0[31:16] | tensor_dim1[15:0]
    g1[3] = (int)((dim1 >> 16) | (tile0 << 16));  // tensor_dim1[31:16] | tile_dim0
    g1[4] = (int)(tile1 & 0xffff);                // tile_dim1 (tile_dim2 = 0)
    g1[5] = (int)stride0;                         // tensor_dim0_stride[31:0]
    g1[6] = 0;
    g1[7] = 0;
    v4i z4 = {0, 0, 0, 0};
    v8i z8 = {0, 0, 0, 0, 0, 0, 0, 0};
    __builtin_amdgcn_tensor_load_to_lds(g0, g1, z4, z4, z8, 0);
}
#endif

// -------- GEMM: C[M,N] = A[M,K] * W[N,K]^T, f32 in/out, bf16 WMMA math --------
__global__ __launch_bounds__(256)
void gemm_bf16wmma(const float* __restrict__ A, const float* __restrict__ W,
                   float* __restrict__ C, int M, int N, int K)
{
    constexpr int BM = 64, BN = 128, BK = 64, PAD = 8;
    __shared__ __align__(16) __bf16 As[BM][BK + PAD];   //  9.2 KB
    __shared__ __align__(16) __bf16 Bs[BN][BK + PAD];   // 18.4 KB

    const int tid  = threadIdx.x;
    const int lane = tid & 31;
    const int wave = tid >> 5;
    const int wm = wave & 1;       // 2 row bands of 32
    const int wn = wave >> 1;      // 4 col bands of 32
    const int row0 = blockIdx.x * BM;
    const int col0 = blockIdx.y * BN;

    const int ar = tid >> 2, ac = (tid & 3) * 16;  // A: 64 x 64, 16 f32/thread
    const int br = tid >> 1, bc = (tid & 1) * 32;  // B: 128 x 64, 32 f32/thread

    const int fr = lane & 15;
    const int fh = (lane >> 4) * 16;

    v8f acc[2][2] = {};

    for (int k0 = 0; k0 < K; k0 += BK) {
        // stage A tile (f32 -> bf16)
        {
            const float4* ap = (const float4*)&A[(size_t)(row0 + ar) * K + k0 + ac];
            st8bf(&As[ar][ac],     ap[0], ap[1]);
            st8bf(&As[ar][ac + 8], ap[2], ap[3]);
            if (k0 + BK < K)
                __builtin_prefetch(&A[(size_t)(row0 + ar) * K + k0 + BK + ac], 0, 3);
        }
        // stage B tile (f32 -> bf16)
        {
            const float4* bp = (const float4*)&W[(size_t)(col0 + br) * K + k0 + bc];
            st8bf(&Bs[br][bc],      bp[0], bp[1]);
            st8bf(&Bs[br][bc + 8],  bp[2], bp[3]);
            st8bf(&Bs[br][bc + 16], bp[4], bp[5]);
            st8bf(&Bs[br][bc + 24], bp[6], bp[7]);
            if (k0 + BK < K)
                __builtin_prefetch(&W[(size_t)(col0 + br) * K + k0 + BK + bc], 0, 3);
        }
        __syncthreads();

#pragma unroll
        for (int kc = 0; kc < 2; ++kc) {
            v16bf af[2], bf[2];
#pragma unroll
            for (int i = 0; i < 2; ++i)
                af[i] = ld_frag(&As[wm * 32 + i * 16 + fr][kc * 32 + fh]);
#pragma unroll
            for (int j = 0; j < 2; ++j)
                bf[j] = ld_frag(&Bs[wn * 32 + j * 16 + fr][kc * 32 + fh]);
#pragma unroll
            for (int i = 0; i < 2; ++i)
#pragma unroll
                for (int j = 0; j < 2; ++j)
                    acc[i][j] = WMMA_BF16(af[i], bf[j], acc[i][j]);
        }
        __syncthreads();
    }

#pragma unroll
    for (int i = 0; i < 2; ++i) {
#pragma unroll
        for (int j = 0; j < 2; ++j) {
            const int mbase = row0 + wm * 32 + i * 16 + ((lane >> 4) << 3);
            const int n     = col0 + wn * 32 + j * 16 + (lane & 15);
#pragma unroll
            for (int r = 0; r < 8; ++r)
                C[(size_t)(mbase + r) * N + n] = acc[i][j][r];
        }
    }
}

// -------- per-head RMSNorm + RoPE, f32 [b,s,NH,D] -> bf16 [b,NH,s,D] --------
__global__ __launch_bounds__(256)
void normrope_kernel(const float* __restrict__ X, const float* __restrict__ nw,
                     const float* __restrict__ cosb, const float* __restrict__ sinb,
                     __bf16* __restrict__ out, int NH)
{
    const int lane = threadIdx.x & 31;
    const int wave = threadIdx.x >> 5;
    const long gid = (long)blockIdx.x * 8 + wave;
    const int h = (int)(gid % NH);
    const long t = gid / NH;
    const int s = (int)(t % Sc);
    const int b = (int)(t / Sc);

    const float* row = X + ((size_t)((size_t)b * Sc + s) * NH + h) * Dc;
    float x0 = row[lane], x1 = row[lane + 32], x2 = row[lane + 64], x3 = row[lane + 96];

    float ss = x0*x0 + x1*x1 + x2*x2 + x3*x3;
#pragma unroll
    for (int off = 16; off >= 1; off >>= 1) ss += __shfl_xor(ss, off, 32);
    const float inv = rsqrtf(ss * (1.0f / 128.0f) + EPSc);

    x0 *= inv * nw[lane];
    x1 *= inv * nw[lane + 32];
    x2 *= inv * nw[lane + 64];
    x3 *= inv * nw[lane + 96];

    const float* cp = cosb + (size_t)((size_t)b * Sc + s) * Dc;
    const float* sp = sinb + (size_t)((size_t)b * Sc + s) * Dc;
    const float c0 = cp[lane], c1 = cp[lane + 32], c2 = cp[lane + 64], c3 = cp[lane + 96];
    const float s0 = sp[lane], s1 = sp[lane + 32], s2 = sp[lane + 64], s3 = sp[lane + 96];

    const float o0 = x0 * c0 - x2 * s0;
    const float o1 = x1 * c1 - x3 * s1;
    const float o2 = x2 * c2 + x0 * s2;
    const float o3 = x3 * c3 + x1 * s3;

    __bf16* op = out + ((size_t)((size_t)b * NH + h) * Sc + s) * Dc;
    op[lane]      = (__bf16)o0;
    op[lane + 32] = (__bf16)o1;
    op[lane + 64] = (__bf16)o2;
    op[lane + 96] = (__bf16)o3;
}

// -------- V: f32 [b,s,kv,d] -> bf16 [b,kv,s,d] --------
__global__ __launch_bounds__(256)
void vconvert_kernel(const float* __restrict__ Vf, __bf16* __restrict__ vb)
{
    const size_t total = (size_t)Bc * Sc * KVc * Dc;
    for (size_t i = (size_t)blockIdx.x * blockDim.x + threadIdx.x; i < total;
         i += (size_t)gridDim.x * blockDim.x) {
        const int d  = (int)(i % Dc);
        size_t t = i / Dc;
        const int kv = (int)(t % KVc); t /= KVc;
        const int s  = (int)(t % Sc);
        const int b  = (int)(t / Sc);
        vb[((size_t)((size_t)b * KVc + kv) * Sc + s) * Dc + d] = (__bf16)Vf[i];
    }
}

// -------- causal GQA flash attention, bf16 WMMA, f32 online softmax --------
__global__ __launch_bounds__(128)
void attn_kernel(const __bf16* __restrict__ qb, const __bf16* __restrict__ kb,
                 const __bf16* __restrict__ vb, float* __restrict__ ctx)
{
    constexpr int T = 64, DP = Dc + 8, TP = T + 8;
    __shared__ __align__(16) __bf16 Qs[T][DP];
    __shared__ __align__(16) __bf16 Ks[T][DP];
    __shared__ __align__(16) __bf16 Vt[Dc][TP];   // transposed V: [d][t]
    __shared__ __align__(16) __bf16 Ps[T][TP];

    const int tid  = threadIdx.x;
    const int lane = tid & 31;
    const int wave = tid >> 5;
    const int qt = blockIdx.x;
    const int bh = blockIdx.y;
    const int h  = bh % Hc;
    const int b  = bh / Hc;
    const int kv = h / (Hc / KVc);

    const __bf16* qbase = qb + ((size_t)((size_t)b * Hc  + h ) * Sc + (size_t)qt * T) * Dc;
    const __bf16* kbase = kb + ((size_t)((size_t)b * KVc + kv) * Sc) * Dc;
    const __bf16* vbase = vb + ((size_t)((size_t)b * KVc + kv) * Sc) * Dc;

#if USE_TDM
    // Q tile via Tensor Data Mover (wave 0 issues, 16B row pad via descriptor)
    if (wave == 0) {
        tdm_load_tile_2d((u32)(uintptr_t)&Qs[0][0], qbase,
                         Dc, Sc, Dc, T, Dc, /*pad 64 DW interval*/5, /*4 DW*/3);
        __builtin_amdgcn_s_wait_tensorcnt(0);
    }
#else
    {
        const int r = tid >> 1, c0 = (tid & 1) * 64;
        const v8bf* src = (const v8bf*)(qbase + (size_t)r * Dc + c0);
#pragma unroll
        for (int j = 0; j < 8; ++j) *(v8bf*)&Qs[r][c0 + j * 8] = src[j];
    }
#endif

    float mstat[8], lsum[8];
    v8f oacc[8] = {};
#pragma unroll
    for (int r = 0; r < 8; ++r) { mstat[r] = -3.0e38f; lsum[r] = 0.0f; }

    const int fr = lane & 15;
    const int fhalf = lane >> 4;
    const int r0 = wave * 16;           // this wave's 16 score rows

    for (int kt = 0; kt <= qt; ++kt) {
#if USE_TDM
        // K tile via TDM (async DMA, overlaps with V staging below)
        if (wave == 0) {
            tdm_load_tile_2d((u32)(uintptr_t)&Ks[0][0],
                             kbase + (size_t)(kt * T) * Dc,
                             Dc, Sc, Dc, T, Dc, 5, 3);
        }
#endif
        // stage transposed V tile (manual: TDM cannot transpose)
        {
            const int r = tid >> 1, c0 = (tid & 1) * 64;
            const v8bf* vsrc = (const v8bf*)(vbase + ((size_t)(kt * T + r)) * Dc + c0);
#if !USE_TDM
            const v8bf* ksrc = (const v8bf*)(kbase + ((size_t)(kt * T + r)) * Dc + c0);
#pragma unroll
            for (int j = 0; j < 8; ++j) *(v8bf*)&Ks[r][c0 + j * 8] = ksrc[j];
#endif
#pragma unroll
            for (int j = 0; j < 8; ++j) {
                v8bf v = vsrc[j];
#pragma unroll
                for (int e = 0; e < 8; ++e) Vt[c0 + j * 8 + e][r] = v[e];
            }
        }
#if USE_TDM
        if (wave == 0) __builtin_amdgcn_s_wait_tensorcnt(0);
#endif
        __syncthreads();

        // S = Q * K^T : 4 col-tiles x 4 K-chunks of 32
        v16bf af[4];
#pragma unroll
        for (int kc = 0; kc < 4; ++kc)
            af[kc] = ld_frag(&Qs[r0 + fr][kc * 32 + fhalf * 16]);

        v8f sacc[4];
#pragma unroll
        for (int nj = 0; nj < 4; ++nj) {
            v8f a = {};
#pragma unroll
            for (int kc = 0; kc < 4; ++kc) {
                v16bf bfg = ld_frag(&Ks[nj * 16 + fr][kc * 32 + fhalf * 16]);
                a = WMMA_BF16(af[kc], bfg, a);
            }
            sacc[nj] = a;
        }

        // scale + causal mask
        const int rowg = qt * T + r0 + fhalf * 8;
#pragma unroll
        for (int nj = 0; nj < 4; ++nj) {
            const int col = kt * T + nj * 16 + fr;
#pragma unroll
            for (int r = 0; r < 8; ++r) {
                float s = sacc[nj][r] * SCALEc;
                if (col > rowg + r) s = -3.0e38f;
                sacc[nj][r] = s;
            }
        }

        // row max (reduce across 16 lanes of each half)
        float mnew[8];
#pragma unroll
        for (int r = 0; r < 8; ++r) {
            float m = mstat[r];
#pragma unroll
            for (int nj = 0; nj < 4; ++nj) m = fmaxf(m, sacc[nj][r]);
#pragma unroll
            for (int off = 8; off >= 1; off >>= 1)
                m = fmaxf(m, __shfl_xor(m, off, 32));
            mnew[r] = m;
        }

        // rescale running state
#pragma unroll
        for (int r = 0; r < 8; ++r) {
            const float alpha = __expf(mstat[r] - mnew[r]);
            lsum[r] *= alpha;
            mstat[r] = mnew[r];
#pragma unroll
            for (int dt = 0; dt < 8; ++dt) oacc[dt][r] *= alpha;
        }

        // P = exp(S - m), stash to LDS as bf16
#pragma unroll
        for (int nj = 0; nj < 4; ++nj) {
#pragma unroll
            for (int r = 0; r < 8; ++r) {
                const float p = __expf(sacc[nj][r] - mstat[r]);
                sacc[nj][r] = p;
                Ps[r0 + fhalf * 8 + r][nj * 16 + fr] = (__bf16)p;
            }
        }

        // row sums
#pragma unroll
        for (int r = 0; r < 8; ++r) {
            float ps = sacc[0][r] + sacc[1][r] + sacc[2][r] + sacc[3][r];
#pragma unroll
            for (int off = 8; off >= 1; off >>= 1) ps += __shfl_xor(ps, off, 32);
            lsum[r] += ps;
        }

        // ctx += P * V : 2 K-chunks of 32 over t, 8 d-tiles
#pragma unroll
        for (int kc = 0; kc < 2; ++kc) {
            v16bf pf = ld_frag(&Ps[r0 + fr][kc * 32 + fhalf * 16]);
#pragma unroll
            for (int dt = 0; dt < 8; ++dt) {
                v16bf vf = ld_frag(&Vt[dt * 16 + fr][kc * 32 + fhalf * 16]);
                oacc[dt] = WMMA_BF16(pf, vf, oacc[dt]);
            }
        }
        __syncthreads();
    }

    // epilogue: ctx[b, s, h, d] = oacc / lsum
#pragma unroll
    for (int dt = 0; dt < 8; ++dt) {
#pragma unroll
        for (int r = 0; r < 8; ++r) {
            const int sg = qt * T + r0 + fhalf * 8 + r;
            const float o = oacc[dt][r] / lsum[r];
            ctx[((size_t)((size_t)b * Sc + sg) * Hc + h) * Dc + dt * 16 + fr] = o;
        }
    }
}

extern "C" void kernel_launch(void* const* d_in, const int* in_sizes, int n_in,
                              void* d_out, int out_size, void* d_ws, size_t ws_size,
                              hipStream_t stream)
{
    const float* hs   = (const float*)d_in[0];
    const float* cosb = (const float*)d_in[1];
    const float* sinb = (const float*)d_in[2];
    const float* q_w  = (const float*)d_in[3];
    const float* k_w  = (const float*)d_in[4];
    const float* v_w  = (const float*)d_in[5];
    const float* o_w  = (const float*)d_in[6];
    const float* qnw  = (const float*)d_in[7];
    const float* knw  = (const float*)d_in[8];

    const size_t M = (size_t)Bc * Sc;      // 4096

    char* ws = (char*)d_ws;
    float*  Qf   = (float*)ws;   ws += M * (size_t)(Hc  * Dc) * 4;
    float*  Kf   = (float*)ws;   ws += M * (size_t)(KVc * Dc) * 4;
    float*  Vf   = (float*)ws;   ws += M * (size_t)(KVc * Dc) * 4;
    __bf16* qbuf = (__bf16*)ws;  ws += M * (size_t)(Hc  * Dc) * 2;
    __bf16* kbuf = (__bf16*)ws;  ws += M * (size_t)(KVc * Dc) * 2;
    __bf16* vbuf = (__bf16*)ws;  ws += M * (size_t)(KVc * Dc) * 2;
    float*  ctx  = (float*)ws;   ws += M * (size_t)(Hc  * Dc) * 4;

    dim3 blk(256);

    // QKV projections
    gemm_bf16wmma<<<dim3((int)M / 64, (Hc  * Dc) / 128), blk, 0, stream>>>(hs, q_w, Qf, (int)M, Hc  * Dc, HIDc);
    gemm_bf16wmma<<<dim3((int)M / 64, (KVc * Dc) / 128), blk, 0, stream>>>(hs, k_w, Kf, (int)M, KVc * Dc, HIDc);
    gemm_bf16wmma<<<dim3((int)M / 64, (KVc * Dc) / 128), blk, 0, stream>>>(hs, v_w, Vf, (int)M, KVc * Dc, HIDc);

    // RMSNorm + RoPE -> bf16, head-major layouts
    normrope_kernel<<<dim3((int)(M * Hc  / 8)), blk, 0, stream>>>(Qf, qnw, cosb, sinb, qbuf, Hc);
    normrope_kernel<<<dim3((int)(M * KVc / 8)), blk, 0, stream>>>(Kf, knw, cosb, sinb, kbuf, KVc);
    vconvert_kernel<<<dim3(4096), blk, 0, stream>>>(Vf, vbuf);

    // causal GQA flash attention
    attn_kernel<<<dim3(Sc / 64, Bc * Hc), dim3(128), 0, stream>>>(qbuf, kbuf, vbuf, ctx);

    // output projection
    gemm_bf16wmma<<<dim3((int)M / 64, HIDc / 128), blk, 0, stream>>>(ctx, o_w, (float*)d_out, (int)M, HIDc, HIDc);
}
